// Qwen2MoTDecoderLayer_2293512536853
// MI455X (gfx1250) — compile-verified, matrix operand order checked by reading, not run on hardware
//
#include <hip/hip_runtime.h>

// ---------------------------------------------------------------------------
// Qwen2-MoT decoder attention layer for MI455X (gfx1250, wave32, WMMA).
// All GEMMs use V_WMMA_F32_16X16X32_BF16 (bf16 in, f32 accumulate).
// x and all weights are pre-converted to bf16 ONCE (memory-bound pass), so
// the GEMM K-loop is pure data movement + WMMA.  GEMM tiles are staged into
// LDS with GLOBAL_LOAD_ASYNC_TO_LDS_B128 (ASYNCcnt) using double buffering.
// Softmax / RMSNorm statistics stay in f32.
// ---------------------------------------------------------------------------

#define T_TOK 8192
#define HID   2048
#define NH    16
#define NKV   4
#define HD    128
#define BB    8
#define LL    1024

typedef unsigned int   u32;
typedef unsigned short u16;
typedef unsigned long long u64;
typedef __attribute__((ext_vector_type(4)))  u32    v4u;
typedef __attribute__((ext_vector_type(8)))  u32    v8u;
typedef __attribute__((ext_vector_type(16))) __bf16 v16bf;
typedef __attribute__((ext_vector_type(8)))  float  v8f;
typedef __attribute__((ext_vector_type(4)))  float  v4f;

#define DEV static __device__ __forceinline__

DEV u16 f2bf(float f) {                 // f32 -> bf16 round-to-nearest-even
  union { float f; u32 u; } x; x.f = f;
  u32 u = x.u;
  u += 0x7FFFu + ((u >> 16) & 1u);
  return (u16)(u >> 16);
}

// Build a 16x32 (bf16) A/B fragment for wave32 WMMA from two 16B chunks:
// lane L<16 holds index (row/col)=L with K {kb..kb+7, kb+16..kb+23}, kb=0;
// lane L>=16 same index, kb=8.  p0 -> K chunk [kb..kb+7], p1 -> [kb+16..kb+23].
DEV v16bf frag16(const u16* p0, const u16* p1) {
  v4u a = *(const v4u*)p0;
  v4u b = *(const v4u*)p1;
  v8u t;
  t[0]=a[0]; t[1]=a[1]; t[2]=a[2]; t[3]=a[3];
  t[4]=b[0]; t[5]=b[1]; t[6]=b[2]; t[7]=b[3];
  return __builtin_bit_cast(v16bf, t);
}

DEV v8f wmma_bf16(v16bf a, v16bf b, v8f c) {
  return __builtin_amdgcn_wmma_f32_16x16x32_bf16(
      false, a, false, b, (short)0, c, false, false);
}

// Async 16B global -> LDS copy (tracked by ASYNCcnt).  The LDS VGPR operand
// is the wave-relative LDS byte offset (low 32 bits of the generic pointer).
DEV void async_cp16(u16* lds_dst, const u16* gsrc) {
  u32 lds_off = (u32)(size_t)lds_dst;
  asm volatile("global_load_async_to_lds_b128 %0, %1, off"
               :: "v"(lds_off), "v"((u64)(size_t)gsrc) : "memory");
}
DEV void wait_async0() {
  asm volatile("s_wait_asynccnt 0" ::: "memory");
}

// ---------------------------------------------------------------------------
// f32 -> bf16 pre-convert (one shot; memory bound). 8 elements / thread.
// ---------------------------------------------------------------------------
__global__ __launch_bounds__(256) void cvt_bf16_kernel(
    const float* __restrict__ in, u16* __restrict__ out)
{
  const int i = (blockIdx.x * 256 + threadIdx.x) * 8;
  v4f a = *(const v4f*)(in + i);
  v4f b = *(const v4f*)(in + i + 4);
  v4u pk;
  pk[0] = (u32)f2bf(a[0]) | ((u32)f2bf(a[1]) << 16);
  pk[1] = (u32)f2bf(a[2]) | ((u32)f2bf(a[3]) << 16);
  pk[2] = (u32)f2bf(b[0]) | ((u32)f2bf(b[1]) << 16);
  pk[3] = (u32)f2bf(b[2]) | ((u32)f2bf(b[3]) << 16);
  *(v4u*)(out + i) = pk;
}

// ---------------------------------------------------------------------------
// Routed GEMM, all-bf16 inputs:
//   out[m,n] = A[m,:] . W{0,1}[n,:] (+ b{0,1}[n]), expert by mask[m]>0.
// Block: 256 thr (8 waves, 4x2 wave grid), tile 64x64, K-step 32.
// Double-buffered async LDS staging; both experts share the A fragment.
// ---------------------------------------------------------------------------
__global__ __launch_bounds__(256) void gemm_routed_bf16(
    const u16* __restrict__ A, const u16* __restrict__ W0,
    const u16* __restrict__ W1, const float* __restrict__ b0,
    const float* __restrict__ b1, const int* __restrict__ mask,
    float* __restrict__ out, int N, int K)
{
  __shared__ __align__(16) u16 sA [2][64 * 32];
  __shared__ __align__(16) u16 sB0[2][64 * 32];
  __shared__ __align__(16) u16 sB1[2][64 * 32];

  const int tid  = threadIdx.x;
  const int lane = tid & 31, wave = tid >> 5;
  const int wm = wave & 3, wn = wave >> 2;          // 4 x 2 waves
  const int tileM = blockIdx.y * 64, tileN = blockIdx.x * 64;
  const int r  = lane & 15;
  const int kb = (lane < 16) ? 0 : 8;

  const int ldRow = tid >> 2;          // 0..63
  const int ldCol = (tid & 3) * 8;     // 0,8,16,24 (u16 elements, 16B chunks)
  const int ldOff = ldRow * 32 + ldCol;

  const u16* aRow  = A  + (size_t)(tileM + ldRow) * K + ldCol;
  const u16* w0Row = W0 + (size_t)(tileN + ldRow) * K + ldCol;
  const u16* w1Row = W1 + (size_t)(tileN + ldRow) * K + ldCol;

  v8f acc[2][2];
  const v8f vzero = {0.f,0.f,0.f,0.f,0.f,0.f,0.f,0.f};
  acc[0][0] = vzero; acc[0][1] = vzero; acc[1][0] = vzero; acc[1][1] = vzero;

  // prologue: stage first K-tile into buffer 0
  async_cp16(&sA [0][ldOff], aRow);
  async_cp16(&sB0[0][ldOff], w0Row);
  async_cp16(&sB1[0][ldOff], w1Row);

  int buf = 0;
  for (int k0 = 0; k0 < K; k0 += 32) {
    wait_async0();                     // own wave's async copies done
    __syncthreads();                   // -> whole tile visible

    if (k0 + 32 < K) {                 // prefetch next tile into other buffer
      const int nk = k0 + 32;
      async_cp16(&sA [buf ^ 1][ldOff], aRow  + nk);
      async_cp16(&sB0[buf ^ 1][ldOff], w0Row + nk);
      async_cp16(&sB1[buf ^ 1][ldOff], w1Row + nk);
    }

    const int ar = (wm * 16 + r) * 32;
    v16bf af = frag16(&sA[buf][ar + kb], &sA[buf][ar + kb + 16]);
#pragma unroll
    for (int nt = 0; nt < 2; ++nt) {
      const int cc = (wn * 32 + nt * 16 + r) * 32;
      v16bf bf0 = frag16(&sB0[buf][cc + kb], &sB0[buf][cc + kb + 16]);
      v16bf bf1 = frag16(&sB1[buf][cc + kb], &sB1[buf][cc + kb + 16]);
      acc[0][nt] = wmma_bf16(af, bf0, acc[0][nt]);
      acc[1][nt] = wmma_bf16(af, bf1, acc[1][nt]);
    }
    __syncthreads();                   // reads done before next overwrite
    buf ^= 1;
  }

  // ---- epilogue: per-row expert select + optional bias ----
  const int rowOff = (lane < 16) ? 0 : 8;
#pragma unroll
  for (int j = 0; j < 8; ++j) {
    const int row = tileM + wm * 16 + j + rowOff;
    const int sel = mask[row] > 0;
#pragma unroll
    for (int nt = 0; nt < 2; ++nt) {
      const int col = tileN + wn * 32 + nt * 16 + r;
      const float bias = b0 ? (sel ? b1[col] : b0[col]) : 0.f;
      out[(size_t)row * N + col] =
          (sel ? acc[1][nt][j] : acc[0][nt][j]) + bias;
    }
  }
}

// ---------------------------------------------------------------------------
// Per-(token,head) RMSNorm (routed weight) + RoPE.  One wave per D=128 row.
// rot_half via cross-lane shuffle (d and d^64 live in lanes L and L^16).
// Output bf16 in (B, NHh, L, D) layout.  `scale` folds 1/sqrt(D) into Q.
// ---------------------------------------------------------------------------
__global__ __launch_bounds__(256) void norm_rope_kernel(
    const float* __restrict__ pre, const float* __restrict__ cosT,
    const float* __restrict__ sinT, const int* __restrict__ mask,
    const float* __restrict__ nw, const float* __restrict__ gnw,
    u16* __restrict__ outp, int NHh, float scale)
{
  const int wave = threadIdx.x >> 5, lane = threadIdx.x & 31;
  const int rowIdx = blockIdx.x * 8 + wave;
  const int t = rowIdx / NHh, h = rowIdx % NHh;
  const int d0 = lane * 4;

  const float* p = pre + (size_t)t * NHh * HD + h * HD + d0;
  v4f x = *(const v4f*)p;
  float ss = x[0]*x[0] + x[1]*x[1] + x[2]*x[2] + x[3]*x[3];
#pragma unroll
  for (int m = 1; m < 32; m <<= 1) ss += __shfl_xor(ss, m, 32);
  const float rs = rsqrtf(ss * (1.0f / HD) + 1e-6f);

  const float* w = (mask[t] > 0) ? gnw : nw;
  const v4f wv = *(const v4f*)(w + d0);
  float y0 = x[0] * rs * wv[0];
  float y1 = x[1] * rs * wv[1];
  float y2 = x[2] * rs * wv[2];
  float y3 = x[3] * rs * wv[3];

  const float sgn = (lane < 16) ? -1.f : 1.f;
  const float q0 = __shfl_xor(y0, 16, 32);
  const float q1 = __shfl_xor(y1, 16, 32);
  const float q2 = __shfl_xor(y2, 16, 32);
  const float q3 = __shfl_xor(y3, 16, 32);

  const v4f cv = *(const v4f*)(cosT + (size_t)t * HD + d0);
  const v4f sv = *(const v4f*)(sinT + (size_t)t * HD + d0);
  const float r0 = (y0 * cv[0] + sgn * q0 * sv[0]) * scale;
  const float r1 = (y1 * cv[1] + sgn * q1 * sv[1]) * scale;
  const float r2 = (y2 * cv[2] + sgn * q2 * sv[2]) * scale;
  const float r3 = (y3 * cv[3] + sgn * q3 * sv[3]) * scale;

  const int b = t / LL, l = t % LL;
  const size_t o = (((size_t)b * NHh + h) * LL + l) * HD + d0;
  uint2 pk;
  pk.x = (u32)f2bf(r0) | ((u32)f2bf(r1) << 16);
  pk.y = (u32)f2bf(r2) | ((u32)f2bf(r3) << 16);
  *(uint2*)(outp + o) = pk;
}

// V: (T, KV*D) f32 -> (B, KV, D, L) bf16 (transposed so P.V B-fragments are
// contiguous along keys).
__global__ __launch_bounds__(256) void vt_kernel(
    const float* __restrict__ pre, u16* __restrict__ vt)
{
  const int idx = blockIdx.x * 256 + threadIdx.x;     // < T*KV*D
  const int t = idx >> 9;                             // / 512
  const int c = idx & 511;
  const int kv = c >> 7, d = c & 127;
  const int b = t >> 10, l = t & 1023;
  vt[((((size_t)b * NKV + kv) * HD + d) << 10) + l] = f2bf(pre[idx]);
}

// ---------------------------------------------------------------------------
// Flash attention, one wave per 16-query tile, 32 keys per step.
// S = Q.K^T -> online softmax in C-layout -> P relayout via per-wave LDS
// (explicit s_wait_dscnt) -> O += P.V with transposed V.
// Causal mask applied analytically (== reference attn_bias).
// ---------------------------------------------------------------------------
__global__ __launch_bounds__(256) void attn_kernel(
    const u16* __restrict__ Q, const u16* __restrict__ Kc,
    const u16* __restrict__ Vt, u16* __restrict__ O)
{
  __shared__ __align__(16) u16 sP[8][16 * 32];

  const int lane = threadIdx.x & 31, wave = threadIdx.x >> 5;
  const int b = blockIdx.z, h = blockIdx.y;
  const int q0 = blockIdx.x * 128 + wave * 16;
  const int kv = h >> 2;                         // H/KV = 4

  const u16* qb = Q  + ((size_t)(b * NH  + h ) * LL) * HD;
  const u16* kp = Kc + ((size_t)(b * NKV + kv) * LL) * HD;
  const u16* vb = Vt + ((size_t)(b * NKV + kv) * HD) * LL;

  const int r  = lane & 15;
  const int kc = (lane < 16) ? 0 : 8;
  const int rowOff = (lane < 16) ? 0 : 8;
  u16* sp = &sP[wave][0];

  v16bf qf[4];
#pragma unroll
  for (int c = 0; c < 4; ++c) {
    const u16* p = qb + (size_t)(q0 + r) * HD + c * 32 + kc;
    qf[c] = frag16(p, p + 16);
  }

  v8f o[8];
  float mr[8], lr[8];
  const v8f vzero = {0.f,0.f,0.f,0.f,0.f,0.f,0.f,0.f};
#pragma unroll
  for (int dt = 0; dt < 8; ++dt) o[dt] = vzero;
#pragma unroll
  for (int j = 0; j < 8; ++j) { mr[j] = -1e30f; lr[j] = 0.f; }

  for (int k0 = 0; k0 < q0 + 16; k0 += 32) {
    v8f s0 = vzero, s1 = vzero;
#pragma unroll
    for (int c = 0; c < 4; ++c) {
      const u16* p0 = kp + (size_t)(k0 + r)      * HD + c * 32 + kc;
      const u16* p1 = kp + (size_t)(k0 + 16 + r) * HD + c * 32 + kc;
      s0 = wmma_bf16(qf[c], frag16(p0, p0 + 16), s0);
      s1 = wmma_bf16(qf[c], frag16(p1, p1 + 16), s1);
    }

#pragma unroll
    for (int j = 0; j < 8; ++j) {
      const int row = q0 + j + rowOff;
      const int c0 = k0 + r, c1 = c0 + 16;
      const float a  = (c0 <= row) ? s0[j] : -1e30f;
      const float bb = (c1 <= row) ? s1[j] : -1e30f;
      float mx = fmaxf(a, bb);
      mx = fmaxf(mx, __shfl_xor(mx, 1, 32));
      mx = fmaxf(mx, __shfl_xor(mx, 2, 32));
      mx = fmaxf(mx, __shfl_xor(mx, 4, 32));
      mx = fmaxf(mx, __shfl_xor(mx, 8, 32));
      const float mnew = fmaxf(mr[j], mx);
      const float al = __expf(mr[j] - mnew);
      const float p0 = __expf(a  - mnew);
      const float p1 = __expf(bb - mnew);
      float rsum = p0 + p1;
      rsum += __shfl_xor(rsum, 1, 32);
      rsum += __shfl_xor(rsum, 2, 32);
      rsum += __shfl_xor(rsum, 4, 32);
      rsum += __shfl_xor(rsum, 8, 32);
      lr[j] = lr[j] * al + rsum;
      mr[j] = mnew;
#pragma unroll
      for (int dt = 0; dt < 8; ++dt) o[dt][j] *= al;
      sp[(j + rowOff) * 32 + r]      = f2bf(p0);
      sp[(j + rowOff) * 32 + r + 16] = f2bf(p1);
    }

    // cross-lane LDS RAW inside the wave: wait for the DS stores.
    asm volatile("s_wait_dscnt 0" ::: "memory");

    v16bf pf = frag16(&sp[r * 32 + kc], &sp[r * 32 + kc + 16]);
#pragma unroll
    for (int dt = 0; dt < 8; ++dt) {
      const u16* vp = vb + (size_t)(dt * 16 + r) * LL + k0 + kc;
      o[dt] = wmma_bf16(pf, frag16(vp, vp + 16), o[dt]);
    }
    // next-iter DS stores stay ordered behind these DS loads (LDS in-order).
  }

#pragma unroll
  for (int j = 0; j < 8; ++j) {
    const float inv = 1.0f / lr[j];
    const int t = b * LL + q0 + j + rowOff;
#pragma unroll
    for (int dt = 0; dt < 8; ++dt)
      O[(size_t)t * (NH * HD) + h * HD + dt * 16 + r] = f2bf(o[dt][j] * inv);
  }
}

// ---------------------------------------------------------------------------
// Launch.  Workspace layout (216 MiB total):
//   [  0.. 64M) pre_q f32   (reused as o_buf bf16 once norm_rope consumes it)
//   [ 64.. 80M) pre_k f32
//   [ 80.. 96M) pre_v f32
//   [ 96..128M) q_attn bf16 (B,H,L,D)
//   [128..136M) k_attn bf16 (B,KV,L,D)
//   [136..144M) v_t    bf16 (B,KV,D,L)
//   [144..176M) xb     bf16 (T,HID)
//   [176..216M) bf16 weights: Wq,Wqg (8M each) Wk,Wkg,Wv,Wvg (2M) Wo,Wog (8M)
// ---------------------------------------------------------------------------
extern "C" void kernel_launch(void* const* d_in, const int* in_sizes, int n_in,
                              void* d_out, int out_size, void* d_ws, size_t ws_size,
                              hipStream_t stream) {
  (void)in_sizes; (void)n_in; (void)out_size; (void)ws_size;
  const float* x    = (const float*)d_in[0];
  const float* cosT = (const float*)d_in[1];
  const float* sinT = (const float*)d_in[2];
  // d_in[3] attn_bias: pure causal mask, applied analytically in attn_kernel.
  const int*   mask = (const int*)d_in[4];
  const float* Wq   = (const float*)d_in[5];
  const float* bq   = (const float*)d_in[6];
  const float* Wqg  = (const float*)d_in[7];
  const float* bqg  = (const float*)d_in[8];
  const float* Wk   = (const float*)d_in[9];
  const float* bk   = (const float*)d_in[10];
  const float* Wkg  = (const float*)d_in[11];
  const float* bkg  = (const float*)d_in[12];
  const float* Wv   = (const float*)d_in[13];
  const float* bv   = (const float*)d_in[14];
  const float* Wvg  = (const float*)d_in[15];
  const float* bvg  = (const float*)d_in[16];
  const float* Wo   = (const float*)d_in[17];
  const float* Wog  = (const float*)d_in[18];
  const float* qnw  = (const float*)d_in[19];
  const float* qgnw = (const float*)d_in[20];
  const float* knw  = (const float*)d_in[21];
  const float* kgnw = (const float*)d_in[22];

  char* ws = (char*)d_ws;
  const size_t MB = 1024 * 1024;
  float* pre_q  = (float*)(ws);
  float* pre_k  = (float*)(ws + 64 * MB);
  float* pre_v  = (float*)(ws + 80 * MB);
  u16*   q_attn = (u16*)(ws + 96 * MB);
  u16*   k_attn = (u16*)(ws + 128 * MB);
  u16*   v_t    = (u16*)(ws + 136 * MB);
  u16*   xb     = (u16*)(ws + 144 * MB);
  u16*   Wq_b   = (u16*)(ws + 176 * MB);
  u16*   Wqg_b  = (u16*)(ws + 184 * MB);
  u16*   Wk_b   = (u16*)(ws + 192 * MB);
  u16*   Wkg_b  = (u16*)(ws + 194 * MB);
  u16*   Wv_b   = (u16*)(ws + 196 * MB);
  u16*   Wvg_b  = (u16*)(ws + 198 * MB);
  u16*   Wo_b   = (u16*)(ws + 200 * MB);
  u16*   Wog_b  = (u16*)(ws + 208 * MB);
  u16*   o_buf  = (u16*)(ws);            // reuses pre_q region

  const dim3 blk(256);
  const int BIG = HID * HID / 2048;      // 2048 blocks for 2048x2048
  const int SML = (NKV * HD) * HID / 2048;

  // 0) one-shot bf16 pre-conversion of activations + weights
  cvt_bf16_kernel<<<dim3(T_TOK * HID / 2048), blk, 0, stream>>>(x,   xb);
  cvt_bf16_kernel<<<dim3(BIG), blk, 0, stream>>>(Wq,  Wq_b);
  cvt_bf16_kernel<<<dim3(BIG), blk, 0, stream>>>(Wqg, Wqg_b);
  cvt_bf16_kernel<<<dim3(SML), blk, 0, stream>>>(Wk,  Wk_b);
  cvt_bf16_kernel<<<dim3(SML), blk, 0, stream>>>(Wkg, Wkg_b);
  cvt_bf16_kernel<<<dim3(SML), blk, 0, stream>>>(Wv,  Wv_b);
  cvt_bf16_kernel<<<dim3(SML), blk, 0, stream>>>(Wvg, Wvg_b);
  cvt_bf16_kernel<<<dim3(BIG), blk, 0, stream>>>(Wo,  Wo_b);
  cvt_bf16_kernel<<<dim3(BIG), blk, 0, stream>>>(Wog, Wog_b);

  // 1) routed QKV projections (bf16 WMMA, dual-expert, f32 out)
  gemm_routed_bf16<<<dim3(HID / 64, T_TOK / 64), blk, 0, stream>>>(
      xb, Wq_b, Wqg_b, bq, bqg, mask, pre_q, NH * HD, HID);
  gemm_routed_bf16<<<dim3((NKV * HD) / 64, T_TOK / 64), blk, 0, stream>>>(
      xb, Wk_b, Wkg_b, bk, bkg, mask, pre_k, NKV * HD, HID);
  gemm_routed_bf16<<<dim3((NKV * HD) / 64, T_TOK / 64), blk, 0, stream>>>(
      xb, Wv_b, Wvg_b, bv, bvg, mask, pre_v, NKV * HD, HID);
  // 2) RMSNorm + RoPE (Q gets 1/sqrt(D) folded in), bf16 out
  norm_rope_kernel<<<dim3((T_TOK * NH) / 8), blk, 0, stream>>>(
      pre_q, cosT, sinT, mask, qnw, qgnw, q_attn, NH, 0.088388347648318447f);
  norm_rope_kernel<<<dim3((T_TOK * NKV) / 8), blk, 0, stream>>>(
      pre_k, cosT, sinT, mask, knw, kgnw, k_attn, NKV, 1.0f);
  // 3) V transpose to (B,KV,D,L) bf16
  vt_kernel<<<dim3((T_TOK * NKV * HD) / 256), blk, 0, stream>>>(pre_v, v_t);
  // 4) causal GQA flash attention
  attn_kernel<<<dim3(LL / 128, NH, BB), blk, 0, stream>>>(
      q_attn, k_attn, v_t, o_buf);
  // 5) routed output projection -> d_out (f32)
  gemm_routed_bf16<<<dim3(HID / 64, T_TOK / 64), blk, 0, stream>>>(
      o_buf, Wo_b, Wog_b, nullptr, nullptr, mask, (float*)d_out, HID, NH * HD);
}
